// GeATLayerWithSingleHead_81200651698825
// MI455X (gfx1250) — matrix operand order repaired and checked by reading, not compiled, not run on hardware
//
#include <hip/hip_runtime.h>
#include <hip/hip_bf16.h>
#include <stdint.h>

// GeAT single-head attention, CDNA5 (gfx1250, wave32) WMMA bf16 pipeline.
// Dominant cost: 137 GF of score GEMMs -> v_wmma_f32_16x16x32_bf16.
// This revision: pipelined B-fragment loads, shared A-fragments across key
// subtiles, async global->LDS staging (ASYNCcnt), and L2 prefetch.

typedef __bf16 bf16t;
typedef __attribute__((ext_vector_type(16))) __bf16 v16bf;
typedef __attribute__((ext_vector_type(8)))  float  v8f;

#define B_ 32
#define N_ 512
#define D_ 128
#define T_ 8
#define NEG_SLOPE 0.2f
#define MASK_VAL (-1e10f)

union FragU { uint4 u[2]; v16bf v; };
union AccU  { v8f v; float f[8]; };

static __device__ __forceinline__ v8f wmma_bf16(const FragU& a, const FragU& b, v8f c) {
  return __builtin_amdgcn_wmma_f32_16x16x32_bf16(false, a.v, false, b.v, (short)0, c, false, false);
}

// A fragment (16xK tile, row-major source, row stride ld elems), K window [k0,k0+32)
// lane<16: M=lane, K = k0+0..15; lane>=16: M=lane-16, K = k0+16..31
static __device__ __forceinline__ FragU load_frag_A(const bf16t* base, int row, int ld, int k0, int lane) {
  int r  = row + (lane & 15);
  int kb = k0 + ((lane & 16) ? 16 : 0);
  const bf16t* p = base + (long)r * ld + kb;
  FragU f;
  f.u[0] = *(const uint4*)(p);
  f.u[1] = *(const uint4*)(p + 8);
  return f;
}

// B fragment: logical B[k][n] stored "n-major" (rows = n, contiguous k, stride ld)
// lane<16: N=lane, K = k0+0..15; lane>=16: N=lane-16, K = k0+16..31
static __device__ __forceinline__ FragU load_frag_B(const bf16t* base, int col, int ld, int k0, int lane) {
  int c  = col + (lane & 15);
  int kb = k0 + ((lane & 16) ? 16 : 0);
  const bf16t* p = base + (long)c * ld + kb;
  FragU f;
  f.u[0] = *(const uint4*)(p);
  f.u[1] = *(const uint4*)(p + 8);
  return f;
}

// Async global -> LDS, 16B per lane (GLOBAL_LOAD_ASYNC_TO_LDS_B128, ASYNCcnt).
static __device__ __forceinline__ void async_g2l_b128(unsigned lds_off, const void* gaddr) {
  asm volatile("global_load_async_to_lds_b128 %0, %1, off" :: "v"(lds_off), "v"(gaddr) : "memory");
}
static __device__ __forceinline__ void wait_async0() {
  asm volatile("s_wait_asynccnt 0x0" ::: "memory");
}
static __device__ __forceinline__ unsigned lds_off(const void* p) {
  // flat LDS-aperture address: low 32 bits are the workgroup-relative LDS offset
  return (unsigned)(size_t)p;
}

__global__ void k_cvt(const float* __restrict__ s, bf16t* __restrict__ d, int n) {
  int i = blockIdx.x * blockDim.x + threadIdx.x;
  if (i < n) d[i] = (bf16t)s[i];
}

// a[d][e][t] (D,D,T) -> aT[t][e][d] so score-weight B-fragments have contiguous K(=d)
__global__ void k_cvt_a(const float* __restrict__ a, bf16t* __restrict__ aT) {
  int i = blockIdx.x * blockDim.x + threadIdx.x;
  if (i >= D_ * D_ * T_) return;
  int d = i % D_;
  int e = (i / D_) % D_;
  int t = i / (D_ * D_);
  aT[i] = (bf16t)a[(d * D_ + e) * T_ + t];
}

// src/dst/val projections: (B*N,128) @ W^T + b.  4 waves/block, 16 rows/wave.
// val is written transposed (b, d, n) for the att@val B-fragments.
__global__ void k_proj(const bf16t* __restrict__ Xbf,
                       const bf16t* __restrict__ Qw, const float* __restrict__ Qb,
                       const bf16t* __restrict__ Kw, const float* __restrict__ Kb,
                       const bf16t* __restrict__ Vw, const float* __restrict__ Vb,
                       bf16t* __restrict__ srcb, bf16t* __restrict__ dstb,
                       bf16t* __restrict__ valT) {
  int lane = threadIdx.x & 31;
  int wave = threadIdx.x >> 5;
  int row0 = blockIdx.x * 64 + wave * 16;
  int l15 = lane & 15, h8 = (lane & 16) ? 8 : 0;
  int bidx = row0 / N_;

  FragU afr[4];
#pragma unroll
  for (int ks = 0; ks < 4; ++ks) afr[ks] = load_frag_A(Xbf, row0, D_, ks * 32, lane);

#pragma unroll
  for (int w = 0; w < 3; ++w) {
    const bf16t* W    = (w == 0) ? Qw : ((w == 1) ? Kw : Vw);
    const float* bias = (w == 0) ? Qb : ((w == 1) ? Kb : Vb);
#pragma unroll
    for (int ct = 0; ct < 8; ++ct) {
      FragU bf[4];  // one load clause, then back-to-back WMMAs
#pragma unroll
      for (int ks = 0; ks < 4; ++ks) bf[ks] = load_frag_B(W, ct * 16, D_, ks * 32, lane);
      AccU acc;
#pragma unroll
      for (int r = 0; r < 8; ++r) acc.f[r] = 0.f;
#pragma unroll
      for (int ks = 0; ks < 4; ++ks) acc.v = wmma_bf16(afr[ks], bf[ks], acc.v);
      int col = ct * 16 + l15;
      float bv = bias[col];
#pragma unroll
      for (int r = 0; r < 8; ++r) {
        int g = row0 + r + h8;
        float v = acc.f[r] + bv;
        if (w == 0)      srcb[(long)g * D_ + col] = (bf16t)v;
        else if (w == 1) dstb[(long)g * D_ + col] = (bf16t)v;
        else             valT[((long)bidx * D_ + col) * N_ + (g - bidx * N_)] = (bf16t)v;
      }
    }
  }
}

// Fused attention: one wave (32 threads) per (batch, 16-row query tile).
// LDS: tslab[t][16][128] (32KB) | dstage[32][128] (8KB) | vstage[128][32] (8KB) | pstage[16][32] (1KB)
__global__ void k_attn(const bf16t* __restrict__ srcb, const bf16t* __restrict__ dstb,
                       const bf16t* __restrict__ valT, const bf16t* __restrict__ aT,
                       const int* __restrict__ edges, bf16t* __restrict__ opre) {
  extern __shared__ char smem[];
  bf16t* tslab  = (bf16t*)(smem);
  bf16t* dstage = (bf16t*)(smem + 32768);
  bf16t* vstage = (bf16t*)(smem + 40960);
  bf16t* pstage = (bf16t*)(smem + 49152);

  int lane = threadIdx.x & 31;
  int l15 = lane & 15, h8 = (lane & 16) ? 8 : 0;
  int b = blockIdx.y;
  int n0 = blockIdx.x * 16;
  int g0 = b * N_ + n0;

  // Kick off async staging of the first key chunk while we compute ts.
  {
    unsigned dso = lds_off(dstage), vso = lds_off(vstage);
    const char* gd = (const char*)(dstb + (long)b * N_ * D_);
    for (int i = lane; i < 512; i += 32) async_g2l_b128(dso + i * 16, gd + i * 16);
    for (int i = lane; i < 512; i += 32) {
      int d = i >> 2, q = i & 3;
      async_g2l_b128(vso + i * 16, (const char*)(valT + ((long)b * D_ + d) * N_) + q * 16);
    }
  }

  // Phase 1: ts tile for all 8 types -> LDS (256 WMMAs). A-frags of src reused across t.
  {
    FragU afr[4];
#pragma unroll
    for (int ks = 0; ks < 4; ++ks) afr[ks] = load_frag_A(srcb, g0, D_, ks * 32, lane);
    for (int t = 0; t < T_; ++t) {
      const bf16t* at = aT + t * D_ * D_;
#pragma unroll
      for (int ct = 0; ct < 8; ++ct) {
        FragU bf[4];
#pragma unroll
        for (int ks = 0; ks < 4; ++ks) bf[ks] = load_frag_B(at, ct * 16, D_, ks * 32, lane);
        AccU acc;
#pragma unroll
        for (int r = 0; r < 8; ++r) acc.f[r] = 0.f;
#pragma unroll
        for (int ks = 0; ks < 4; ++ks) acc.v = wmma_bf16(afr[ks], bf[ks], acc.v);
#pragma unroll
        for (int r = 0; r < 8; ++r)
          tslab[(t * 16 + r + h8) * D_ + ct * 16 + l15] = (bf16t)acc.f[r];
      }
    }
  }

  float m_run[8], l_run[8];
  AccU oacc[8];
#pragma unroll
  for (int r = 0; r < 8; ++r) { m_run[r] = -3e38f; l_run[r] = 0.f; }
#pragma unroll
  for (int dt = 0; dt < 8; ++dt)
#pragma unroll
    for (int r = 0; r < 8; ++r) oacc[dt].f[r] = 0.f;

  for (int mc = 0; mc < 16; ++mc) {
    int m0 = mc * 32;
    wait_async0();  // chunk mc resident in dstage/vstage

    // L2 prefetch of next chunk (global_prefetch_b8), then consume current.
    if (mc + 1 < 16) {
      const char* pd = (const char*)(dstb + ((long)b * N_ + m0 + 32) * D_);
      __builtin_prefetch(pd + lane * 256, 0, 1);
      __builtin_prefetch(pd + lane * 256 + 128, 0, 1);
      __builtin_prefetch(edges + ((long)b * N_ + n0 + (lane & 15)) * N_ + m0 + 32, 0, 1);
    }

    // Both 16-key subtiles' dst B-fragments resident: one A load feeds two WMMAs.
    FragU bfr[2][4];
#pragma unroll
    for (int sub = 0; sub < 2; ++sub)
#pragma unroll
      for (int ks = 0; ks < 4; ++ks) bfr[sub][ks] = load_frag_B(dstage, sub * 16, D_, ks * 32, lane);

    int et[2][8];
#pragma unroll
    for (int sub = 0; sub < 2; ++sub)
#pragma unroll
      for (int r = 0; r < 8; ++r)
        et[sub][r] = edges[((long)b * N_ + (n0 + r + h8)) * N_ + m0 + sub * 16 + l15];

    AccU sel0, sel1;
#pragma unroll
    for (int r = 0; r < 8; ++r) { sel0.f[r] = 0.f; sel1.f[r] = 0.f; }
#pragma unroll
    for (int t = 0; t < T_; ++t) {
      AccU a0, a1;
#pragma unroll
      for (int r = 0; r < 8; ++r) { a0.f[r] = 0.f; a1.f[r] = 0.f; }
#pragma unroll
      for (int ks = 0; ks < 4; ++ks) {
        FragU af = load_frag_A(tslab + t * 16 * D_, 0, D_, ks * 32, lane);
        a0.v = wmma_bf16(af, bfr[0][ks], a0.v);
        a1.v = wmma_bf16(af, bfr[1][ks], a1.v);
      }
#pragma unroll
      for (int r = 0; r < 8; ++r) {
        if ((et[0][r] < 0 ? 0 : et[0][r]) == t) sel0.f[r] = a0.f[r];  // clamp(-1 -> 0)
        if ((et[1][r] < 0 ? 0 : et[1][r]) == t) sel1.f[r] = a1.f[r];
      }
    }

    float sv[2][8];
#pragma unroll
    for (int r = 0; r < 8; ++r) {
      float v0 = (et[0][r] == -1) ? MASK_VAL : sel0.f[r];
      float v1 = (et[1][r] == -1) ? MASK_VAL : sel1.f[r];
      sv[0][r] = (v0 >= 0.f) ? v0 : NEG_SLOPE * v0;  // leaky relu (mask included, like ref)
      sv[1][r] = (v1 >= 0.f) ? v1 : NEG_SLOPE * v1;
    }

    // Online softmax: row n = r + h8 spans the 16 lanes of a half-wave -> xor 1/2/4/8.
    float pr[2][8];
#pragma unroll
    for (int r = 0; r < 8; ++r) {
      float cm = fmaxf(sv[0][r], sv[1][r]);
      cm = fmaxf(cm, __shfl_xor(cm, 1));
      cm = fmaxf(cm, __shfl_xor(cm, 2));
      cm = fmaxf(cm, __shfl_xor(cm, 4));
      cm = fmaxf(cm, __shfl_xor(cm, 8));
      float nm = fmaxf(m_run[r], cm);
      float sc = __expf(m_run[r] - nm);
      float p0 = __expf(sv[0][r] - nm);
      float p1 = __expf(sv[1][r] - nm);
      float rs = p0 + p1;
      rs += __shfl_xor(rs, 1);
      rs += __shfl_xor(rs, 2);
      rs += __shfl_xor(rs, 4);
      rs += __shfl_xor(rs, 8);
      l_run[r] = l_run[r] * sc + rs;
      m_run[r] = nm;
      pr[0][r] = p0; pr[1][r] = p1;
#pragma unroll
      for (int dt = 0; dt < 8; ++dt) oacc[dt].f[r] *= sc;
    }

    // C-fragment -> A-fragment relayout of P through LDS, then att@val (8 WMMAs).
#pragma unroll
    for (int sub = 0; sub < 2; ++sub)
#pragma unroll
      for (int r = 0; r < 8; ++r)
        pstage[(r + h8) * 32 + sub * 16 + l15] = (bf16t)pr[sub][r];

    FragU ap = load_frag_A(pstage, 0, 32, 0, lane);
    FragU vb[4];  // only 8 tiles; load in two clauses of 4
#pragma unroll
    for (int dt = 0; dt < 4; ++dt) vb[dt] = load_frag_B(vstage, dt * 16, 32, 0, lane);
#pragma unroll
    for (int dt = 0; dt < 4; ++dt) oacc[dt].v = wmma_bf16(ap, vb[dt], oacc[dt].v);
#pragma unroll
    for (int dt = 0; dt < 4; ++dt) vb[dt] = load_frag_B(vstage, (dt + 4) * 16, 32, 0, lane);
#pragma unroll
    for (int dt = 0; dt < 4; ++dt) oacc[dt + 4].v = wmma_bf16(ap, vb[dt], oacc[dt + 4].v);

    // All LDS reads of this chunk are consumed (WMMAs waited on DScnt):
    // safe to issue next chunk's async stores into dstage/vstage.
    if (mc + 1 < 16) {
      int m1 = m0 + 32;
      unsigned dso = lds_off(dstage), vso = lds_off(vstage);
      const char* gd = (const char*)(dstb + ((long)b * N_ + m1) * D_);
      for (int i = lane; i < 512; i += 32) async_g2l_b128(dso + i * 16, gd + i * 16);
      for (int i = lane; i < 512; i += 32) {
        int d = i >> 2, q = i & 3;
        async_g2l_b128(vso + i * 16, (const char*)(valT + ((long)b * D_ + d) * N_ + m1) + q * 16);
      }
    }
  }

  // Normalize and store pre-projection output (bf16, row-major) for the final GEMM.
#pragma unroll
  for (int dt = 0; dt < 8; ++dt) {
    int d = dt * 16 + l15;
#pragma unroll
    for (int r = 0; r < 8; ++r) {
      float v = oacc[dt].f[r] / l_run[r];
      opre[(long)(g0 + r + h8) * D_ + d] = (bf16t)v;
    }
  }
}

// out = opre @ Pw^T + Pb (fp32 result)
__global__ void k_final(const bf16t* __restrict__ opre, const bf16t* __restrict__ Pw,
                        const float* __restrict__ Pb, float* __restrict__ out) {
  int lane = threadIdx.x & 31;
  int wave = threadIdx.x >> 5;
  int row0 = blockIdx.x * 64 + wave * 16;
  int l15 = lane & 15, h8 = (lane & 16) ? 8 : 0;
  FragU afr[4];
#pragma unroll
  for (int ks = 0; ks < 4; ++ks) afr[ks] = load_frag_A(opre, row0, D_, ks * 32, lane);
#pragma unroll
  for (int ct = 0; ct < 8; ++ct) {
    FragU bf[4];
#pragma unroll
    for (int ks = 0; ks < 4; ++ks) bf[ks] = load_frag_B(Pw, ct * 16, D_, ks * 32, lane);
    AccU acc;
#pragma unroll
    for (int r = 0; r < 8; ++r) acc.f[r] = 0.f;
#pragma unroll
    for (int ks = 0; ks < 4; ++ks) acc.v = wmma_bf16(afr[ks], bf[ks], acc.v);
    int col = ct * 16 + l15;
    float bv = Pb[col];
#pragma unroll
    for (int r = 0; r < 8; ++r)
      out[(long)(row0 + r + h8) * D_ + col] = acc.f[r] + bv;
  }
}

extern "C" void kernel_launch(void* const* d_in, const int* in_sizes, int n_in,
                              void* d_out, int out_size, void* d_ws, size_t ws_size,
                              hipStream_t stream) {
  const float* X     = (const float*)d_in[0];
  const int*   edges = (const int*)d_in[1];
  const float* Qw = (const float*)d_in[2];
  const float* Qb = (const float*)d_in[3];
  const float* Kw = (const float*)d_in[4];
  const float* Kb = (const float*)d_in[5];
  const float* Vw = (const float*)d_in[6];
  const float* Vb = (const float*)d_in[7];
  const float* a  = (const float*)d_in[8];
  const float* Pw = (const float*)d_in[9];
  const float* Pb = (const float*)d_in[10];
  float* out = (float*)d_out;

  char* ws = (char*)d_ws;
  bf16t* Xbf  = (bf16t*)(ws + 0);         // 4 MB
  bf16t* srcb = (bf16t*)(ws + 4194304);   // 4 MB
  bf16t* dstb = (bf16t*)(ws + 8388608);   // 4 MB
  bf16t* valT = (bf16t*)(ws + 12582912);  // 4 MB
  bf16t* opre = (bf16t*)(ws + 16777216);  // 4 MB
  bf16t* Qwb  = (bf16t*)(ws + 20971520);  // 32 KB
  bf16t* Kwb  = (bf16t*)(ws + 21004288);
  bf16t* Vwb  = (bf16t*)(ws + 21037056);
  bf16t* Pwb  = (bf16t*)(ws + 21069824);
  bf16t* aTb  = (bf16t*)(ws + 21102592);  // 256 KB -> total ~20.4 MB

  k_cvt<<<(B_ * N_ * D_ + 255) / 256, 256, 0, stream>>>(X, Xbf, B_ * N_ * D_);
  k_cvt<<<(D_ * D_ + 255) / 256, 256, 0, stream>>>(Qw, Qwb, D_ * D_);
  k_cvt<<<(D_ * D_ + 255) / 256, 256, 0, stream>>>(Kw, Kwb, D_ * D_);
  k_cvt<<<(D_ * D_ + 255) / 256, 256, 0, stream>>>(Vw, Vwb, D_ * D_);
  k_cvt<<<(D_ * D_ + 255) / 256, 256, 0, stream>>>(Pw, Pwb, D_ * D_);
  k_cvt_a<<<(D_ * D_ * T_ + 255) / 256, 256, 0, stream>>>(a, aTb);

  k_proj<<<(B_ * N_) / 64, 128, 0, stream>>>(Xbf, Qwb, Qb, Kwb, Kb, Vwb, Vb, srcb, dstb, valT);
  k_attn<<<dim3(N_ / 16, B_), 32, 50176, stream>>>(srcb, dstb, valT, aTb, edges, opre);
  k_final<<<(B_ * N_) / 64, 128, 0, stream>>>(opre, Pwb, Pb, out);
}
